// InfluenceGraphConv_70136815944295
// MI455X (gfx1250) — compile-verified, matrix operand order checked by reading, not run on hardware
//
#include <hip/hip_runtime.h>

// CDNA5 / gfx1250 — wave32, WMMA 16x16x32 f16->f32 path.
typedef __attribute__((ext_vector_type(16))) _Float16 v16h;
typedef __attribute__((ext_vector_type(8)))  float    v8f;

#define K_DIM      128
#define N_OUT      32
#define LDS_STRIDE 130   // 128 halves + 2-half pad: lane stride 260B -> rotates LDS bank per lane

// -------------------- Kernel 0: zero the output accumulator --------------------
__global__ void zero_out_f4(float4* __restrict__ out, int n4) {
    int i = blockIdx.x * blockDim.x + threadIdx.x;
    if (i < n4) out[i] = make_float4(0.f, 0.f, 0.f, 0.f);
}

// -------------------- Kernel 1: h = feat @ W via v_wmma_f32_16x16x32_f16 --------------------
// Block = 256 threads = 8 waves. Each wave owns a 16-row tile and the full 32-col output
// (two 16x16 f32 accumulators). K=128 processed as 4 WMMA steps of K=32.
__global__ void __launch_bounds__(256)
gemm_feat_W(const float* __restrict__ feat, const float* __restrict__ W,
            float* __restrict__ h, int n_nodes) {
    __shared__ _Float16 sWt[N_OUT * LDS_STRIDE];   // W transposed, f16, padded (~8.3 KB)

    const int tid = threadIdx.x;
    // Stage W[k][n] -> sWt[n][k] as f16 so each lane's K-run is contiguous in LDS.
    for (int i = tid; i < K_DIM * N_OUT; i += 256) {
        const int k = i / N_OUT;
        const int n = i % N_OUT;
        sWt[n * LDS_STRIDE + k] = (_Float16)W[i];
    }
    __syncthreads();

    const int wave = tid >> 5;
    const int lane = tid & 31;
    const int row0 = (blockIdx.x * 8 + wave) * 16;
    if (row0 >= n_nodes) return;   // wave-uniform: EXEC all-1s inside surviving waves

    // 16-bit A-matrix VGPR layout (ISA 7.12.2): lanes 0-15 -> row M=lane, K pairs
    // {0,1..6,7} in V0..V3 and {16..23} in V4..V7; lanes 16-31 -> same row, K+8.
    const int m  = lane & 15;
    const int kb = (lane < 16) ? 0 : 8;

    const float*    arow  = feat + (size_t)(row0 + m) * K_DIM;
    const _Float16* bcol0 = &sWt[(m)      * LDS_STRIDE];
    const _Float16* bcol1 = &sWt[(m + 16) * LDS_STRIDE];

    v8f acc0 = {};   // output cols 0..15
    v8f acc1 = {};   // output cols 16..31

    #pragma unroll
    for (int k0 = 0; k0 < K_DIM; k0 += 32) {
        v16h a, b0, b1;
        #pragma unroll
        for (int v = 0; v < 8; ++v) {
            const int k = k0 + kb + ((v < 4) ? (2 * v) : (16 + 2 * (v - 4)));
            a[2 * v]     = (_Float16)arow[k];
            a[2 * v + 1] = (_Float16)arow[k + 1];
            b0[2 * v]     = bcol0[k];
            b0[2 * v + 1] = bcol0[k + 1];
            b1[2 * v]     = bcol1[k];
            b1[2 * v + 1] = bcol1[k + 1];
        }
        // (neg_a, A, neg_b, B, c_mod, C, reuse_a, reuse_b)
        acc0 = __builtin_amdgcn_wmma_f32_16x16x32_f16(false, a, false, b0,
                                                      (short)0, acc0, false, false);
        acc1 = __builtin_amdgcn_wmma_f32_16x16x32_f16(false, a, false, b1,
                                                      (short)0, acc1, false, false);
    }

    // 32-bit C/D layout: VGPR r holds (M = r, N = lane) for lanes 0-15 and
    // (M = r+8, N = lane-16) for lanes 16-31.
    const int rbase = row0 + ((lane < 16) ? 0 : 8);
    #pragma unroll
    for (int r = 0; r < 8; ++r) {
        float* orow = h + (size_t)(rbase + r) * N_OUT;
        orow[m]      = acc0[r];
        orow[16 + m] = acc1[r];
    }
}

// -------------------- Kernel 2: out[dst] += h[src] * edge_w --------------------
// 8 threads per edge, float4 gather (L2-resident: h is 12.8 MB << 192 MB L2),
// 4 global_atomic_add_f32 per thread into the L2-resident output.
__global__ void __launch_bounds__(256)
edge_scatter(const float* __restrict__ h, const float* __restrict__ ew,
             const int* __restrict__ src, const int* __restrict__ dst,
             float* __restrict__ out, int n_edges) {
    const long long t = (long long)blockIdx.x * blockDim.x + threadIdx.x;
    const int e = (int)(t >> 3);
    if (e >= n_edges) return;
    const int f = (int)(t & 7) * 4;

    const int   s = src[e];
    const int   d = dst[e];
    const float w = ew[e];

    const float4 hv = *(const float4*)(h + (size_t)s * N_OUT + f);
    float* o = out + (size_t)d * N_OUT + f;
    atomicAdd(o + 0, hv.x * w);
    atomicAdd(o + 1, hv.y * w);
    atomicAdd(o + 2, hv.z * w);
    atomicAdd(o + 3, hv.w * w);
}

// -------------------- Launch --------------------
extern "C" void kernel_launch(void* const* d_in, const int* in_sizes, int n_in,
                              void* d_out, int out_size, void* d_ws, size_t ws_size,
                              hipStream_t stream) {
    const float* feat = (const float*)d_in[0];   // [N,128] f32
    const float* W    = (const float*)d_in[1];   // [128,32] f32
    const float* ew   = (const float*)d_in[2];   // [E] f32
    const int*   src  = (const int*)  d_in[3];   // [E] i32
    const int*   dst  = (const int*)  d_in[4];   // [E] i32
    float*       out  = (float*)d_out;           // [N,32] f32
    float*       h    = (float*)d_ws;            // scratch: [N,32] f32 (12.8 MB)

    const int n_nodes = in_sizes[0] / K_DIM;
    const int n_edges = in_sizes[2];

    // 0) zero output (harness poisons d_out; out_size = N*32 is divisible by 4)
    const int n4 = out_size / 4;
    zero_out_f4<<<(n4 + 255) / 256, 256, 0, stream>>>((float4*)out, n4);

    // 1) WMMA GEMM: h = feat @ W   (N=100000 is a multiple of 16 -> clean tiles)
    const int row_tiles = (n_nodes + 15) / 16;
    const int gblocks   = (row_tiles + 7) / 8;   // 8 waves per block
    gemm_feat_W<<<gblocks, 256, 0, stream>>>(feat, W, h, n_nodes);

    // 2) edge gather/scale/scatter-add
    const long long threads = (long long)n_edges * 8;
    const int sblocks = (int)((threads + 255) / 256);
    edge_scatter<<<sblocks, 256, 0, stream>>>(h, ew, src, dst, out, n_edges);
}
